// EfficientGlobalPointer_62740882260564
// MI455X (gfx1250) — matrix operand order repaired and checked by reading, not compiled, not run on hardware
//
#include <hip/hip_runtime.h>
#include <hip/hip_bf16.h>
#include <stdint.h>

// Problem constants
#define BB   8
#define NN   1024
#define HH   768
#define NHEADS 12
#define HD   64        // head size
#define XC   128       // 2*HD  (columns of x)
#define JC   24        // 2*NHEADS (columns of bias GEMM)
#define NEGC 1000000000000.0f

typedef __attribute__((ext_vector_type(16))) __bf16 v16bf;
typedef __attribute__((ext_vector_type(8)))  __bf16 v8bf;
typedef __attribute__((ext_vector_type(8)))  float  v8f;
typedef __attribute__((ext_vector_type(4)))  float  v4f;

static __device__ __forceinline__ __bf16 to_bf16(float f) {
  unsigned u = __builtin_bit_cast(unsigned, f);
  u += 0x7FFFu + ((u >> 16) & 1u);           // round-to-nearest-even
  unsigned short h = (unsigned short)(u >> 16);
  return __builtin_bit_cast(__bf16, h);
}

// ---------------------------------------------------------------------------
// K0: convert inputs f32->bf16 (row-major) and W1 -> bf16 transposed (N x K)
// ---------------------------------------------------------------------------
__global__ void k_convert(const float* __restrict__ inp, const float* __restrict__ W1,
                          __bf16* __restrict__ inbf, __bf16* __restrict__ w1t) {
  int i = blockIdx.x * 256 + threadIdx.x;
  if (i < BB * NN * HH) inbf[i] = to_bf16(inp[i]);
  if (i < HH * XC) {                 // w1t[n][k] = W1[k][n]
    int n = i / HH, k = i % HH;
    w1t[i] = to_bf16(W1[k * XC + n]);
  }
}

// ---------------------------------------------------------------------------
// K1: x = inputs @ W1 + b1   via WMMA bf16 16x16x32, K=768 (24 steps)
// grid = 512 blocks (m-tiles), 8 waves/block (n-tiles)
// A layout (16x32 bf16): lane<16 holds K 0..15 of row lane;
//                        lane>=16 holds K 16..31.
// B layout (32x16 bf16): lane holds 16 contiguous K of column lane%16,
//                        K half selected by lane/16.
// ---------------------------------------------------------------------------
__global__ void __launch_bounds__(256) k_gemm1(const __bf16* __restrict__ inbf,
                                               const __bf16* __restrict__ w1t,
                                               const float* __restrict__ b1,
                                               float* __restrict__ xbuf) {
  int wave = threadIdx.x >> 5;
  int lane = threadIdx.x & 31;
  int mt = blockIdx.x;               // 0..511
  int nt = wave;                     // 0..7
  int hl = lane >> 4;
  int l15 = lane & 15;

  const __bf16* ap = inbf + (size_t)(mt * 16 + l15) * HH + hl * 16;
  const __bf16* bp = w1t  + (size_t)(nt * 16 + l15) * HH + hl * 16;

  v8f acc = {};
  for (int k0 = 0; k0 < HH; k0 += 32) {
    __builtin_prefetch(ap + 64, 0, 0);       // -> global_prefetch_b8
    v8bf alo = *(const v8bf*)(ap);
    v8bf ahi = *(const v8bf*)(ap + 8);
    v16bf a, b;
    b = *(const v16bf*)(bp);
#pragma unroll
    for (int i = 0; i < 8; ++i) { a[i] = alo[i]; a[i + 8] = ahi[i]; }
    acc = __builtin_amdgcn_wmma_f32_16x16x32_bf16(false, a, false, b,
                                                  (short)0, acc, false, false);
    ap += 32; bp += 32;
  }
  int n = nt * 16 + l15;
  float bias = b1[n];
#pragma unroll
  for (int v = 0; v < 8; ++v) {
    int m = mt * 16 + v + 8 * hl;            // C layout: VGPR v, lane half
    xbuf[(size_t)m * XC + n] = acc[v] + bias;
  }
}

// ---------------------------------------------------------------------------
// K2: RoPE: qw = x[..., ::2], kw = x[..., 1::2]; rotate pairs; emit bf16
// one thread per (row, j<32) rotation pair
// ---------------------------------------------------------------------------
__global__ void k_rope(const float* __restrict__ xbuf,
                       __bf16* __restrict__ qbf, __bf16* __restrict__ kbf) {
  int i = blockIdx.x * 256 + threadIdx.x;    // 8192*32 threads
  int row = i >> 5;
  int j = i & 31;
  int pos = row & (NN - 1);
  const float* xr = xbuf + (size_t)row * XC;
  float inv = __powf(10000.0f, -(float)j / 32.0f);
  float ang = (float)pos * inv;
  float s, c;
  __sincosf(ang, &s, &c);
  float q0 = xr[4 * j],     q1 = xr[4 * j + 2];
  float k0 = xr[4 * j + 1], k1 = xr[4 * j + 3];
  qbf[(size_t)row * HD + 2 * j]     = to_bf16(q0 * c - q1 * s);
  qbf[(size_t)row * HD + 2 * j + 1] = to_bf16(q1 * c + q0 * s);
  kbf[(size_t)row * HD + 2 * j]     = to_bf16(k0 * c - k1 * s);
  kbf[(size_t)row * HD + 2 * j + 1] = to_bf16(k1 * c + k0 * s);
}

// ---------------------------------------------------------------------------
// K3: bias = (x @ W2 + b2)/2, split even cols -> biasN (added along n),
//     odd cols -> biasM (added along m). One thread per (row, j<24).
// ---------------------------------------------------------------------------
__global__ void k_bias(const float* __restrict__ xbuf, const float* __restrict__ W2,
                       const float* __restrict__ b2,
                       float* __restrict__ biasN, float* __restrict__ biasM) {
  int i = blockIdx.x * 256 + threadIdx.x;    // 8192*24 threads (exact)
  int row = i / JC;
  int j = i % JC;
  const float* xr = xbuf + (size_t)row * XC;
  float s = b2[j];
#pragma unroll 4
  for (int kk = 0; kk < XC; ++kk) s += xr[kk] * W2[kk * JC + j];
  s *= 0.5f;
  int b = row >> 10, n = row & (NN - 1);
  int h = j >> 1;
  float* dst = (j & 1) ? biasM : biasN;
  dst[(size_t)(b * NHEADS + h) * NN + n] = s;
}

// ---------------------------------------------------------------------------
// K4: main kernel. Each wave: one 16x32 base tile of qw@kw^T (4 WMMAs),
// stage through LDS (stride 144B, conflict-free), read back as 4x(row,16B),
// fold scale + causal/pad mask once, then loop 12 heads adding biases and
// streaming out with non-temporal b128 stores (4 rows x 128B per store).
// grid = 2048 blocks x 256 threads = 16384 waves = 8 * 64 * 32 tiles.
// ---------------------------------------------------------------------------
__global__ void __launch_bounds__(256) k_logits(const __bf16* __restrict__ qbf,
                                                const __bf16* __restrict__ kbf,
                                                const float* __restrict__ biasN,
                                                const float* __restrict__ biasM,
                                                const int* __restrict__ amask,
                                                float* __restrict__ out) {
  __shared__ __align__(16) char smem[8 * 2304];
  int wib  = threadIdx.x >> 5;
  int lane = threadIdx.x & 31;
  int waveId = blockIdx.x * 8 + wib;
  int b   = waveId >> 11;                    // / 2048
  int rem = waveId & 2047;
  int mt = rem >> 5, nt = rem & 31;
  int m0 = mt * 16, n0 = nt * 32;
  int hl = lane >> 4, l15 = lane & 15;

  const __bf16* ap  = qbf + (size_t)(b * NN + m0 + l15) * HD + hl * 16;
  const __bf16* bp0 = kbf + (size_t)(b * NN + n0 + l15) * HD + hl * 16;
  const __bf16* bp1 = bp0 + 16 * HD;

  v8f acc0 = {}, acc1 = {};
#pragma unroll
  for (int ks = 0; ks < 2; ++ks) {           // K = 64 in 2 steps of 32
    v8bf alo = *(const v8bf*)(ap + ks * 32);
    v8bf ahi = *(const v8bf*)(ap + ks * 32 + 8);
    v16bf a;
#pragma unroll
    for (int i = 0; i < 8; ++i) { a[i] = alo[i]; a[i + 8] = ahi[i]; }
    v16bf bb0 = *(const v16bf*)(bp0 + ks * 32);
    v16bf bb1 = *(const v16bf*)(bp1 + ks * 32);
    acc0 = __builtin_amdgcn_wmma_f32_16x16x32_bf16(false, a, false, bb0,
                                                   (short)0, acc0, false, false);
    acc1 = __builtin_amdgcn_wmma_f32_16x16x32_bf16(false, a, false, bb1,
                                                   (short)0, acc1, false, false);
  }

  // --- stage 16x32 f32 tile to LDS (row stride 144B avoids bank conflicts) --
  char* base = smem + wib * 2304;
#pragma unroll
  for (int v = 0; v < 8; ++v) {
    int row = v + 8 * hl;
    *(float*)(base + row * 144 + l15 * 4)      = acc0[v];
    *(float*)(base + row * 144 + 64 + l15 * 4) = acc1[v];
  }

  // --- read back: lane covers chunk (k*32+lane) -> (row, 16B col chunk) -----
  const float scale = 0.125f;                // 1/sqrt(64)
  v4f bse[4];
  int mrow[4], ncol[4];
#pragma unroll
  for (int k = 0; k < 4; ++k) {
    int flat = k * 32 + lane;
    int rr = flat >> 3, cc = flat & 7;
    v4f r = *(const v4f*)(base + rr * 144 + cc * 16);
    int m  = m0 + rr;
    int nb = n0 + cc * 4;
    mrow[k] = m; ncol[k] = nb;
    float am_m = (float)amask[b * NN + m];
    v4f msk;
#pragma unroll
    for (int e = 0; e < 4; ++e) {
      int n = nb + e;
      float am_n = (float)amask[b * NN + n];
      float pen = -(1.0f - am_m * am_n) * NEGC;   // padding mask
      if (n < m) pen -= NEGC;                     // strict-lower causal mask
      msk[e] = pen;
    }
    bse[k] = r * scale + msk;                // head-invariant part, folded once
  }

  // --- head loop: add per-head biases, stream out non-temporally ------------
  for (int h = 0; h < NHEADS; ++h) {
    const float* bNh = biasN + (size_t)(b * NHEADS + h) * NN;
    const float* bMh = biasM + (size_t)(b * NHEADS + h) * NN;
    float* oph = out + (size_t)(b * NHEADS + h) * NN * NN;
#pragma unroll
    for (int k = 0; k < 4; ++k) {
      v4f bn = *(const v4f*)(bNh + ncol[k]);
      float bm = bMh[mrow[k]];
      v4f val = bse[k] + bn + bm;
      __builtin_nontemporal_store(val, (v4f*)(oph + (size_t)mrow[k] * NN + ncol[k]));
    }
  }
}

// ---------------------------------------------------------------------------
// Host side
// ---------------------------------------------------------------------------
extern "C" void kernel_launch(void* const* d_in, const int* in_sizes, int n_in,
                              void* d_out, int out_size, void* d_ws, size_t ws_size,
                              hipStream_t stream) {
  const float* inputs = (const float*)d_in[0];
  const int*   amask  = (const int*)  d_in[1];
  const float* W1     = (const float*)d_in[2];
  const float* b1     = (const float*)d_in[3];
  const float* W2     = (const float*)d_in[4];
  const float* b2     = (const float*)d_in[5];
  float* out = (float*)d_out;

  // workspace carve-up (~20 MB total; all 16B aligned)
  char* ws = (char*)d_ws;
  __bf16* inbf = (__bf16*)ws;                        // 8192*768 bf16
  __bf16* w1t  = inbf + (size_t)BB * NN * HH;        // 128*768 bf16
  float*  xbuf = (float*)(w1t + (size_t)XC * HH);    // 8192*128 f32
  __bf16* qbf  = (__bf16*)(xbuf + (size_t)BB * NN * XC);
  __bf16* kbf  = qbf + (size_t)BB * NN * HD;
  float* biasN = (float*)(kbf + (size_t)BB * NN * HD);
  float* biasM = biasN + (size_t)BB * NHEADS * NN;

  (void)in_sizes; (void)n_in; (void)out_size; (void)ws_size;

  // K0: convert (6,291,456 dominant elements)
  k_convert<<<(BB * NN * HH + 255) / 256, 256, 0, stream>>>(inputs, W1, inbf, w1t);
  // K1: x = inputs @ W1 + b1 (WMMA)
  k_gemm1<<<512, 256, 0, stream>>>(inbf, w1t, b1, xbuf);
  // K2: RoPE -> bf16 q/k
  k_rope<<<(BB * NN * 32) / 256, 256, 0, stream>>>(xbuf, qbf, kbf);
  // K3: bias GEMM (x @ W2 + b2)/2 split into biasN/biasM
  k_bias<<<(BB * NN * JC) / 256, 256, 0, stream>>>(xbuf, W2, b2, biasN, biasM);
  // K4: logits (WMMA + LDS swizzle + NT streaming stores)
  k_logits<<<2048, 256, 0, stream>>>(qbf, kbf, biasN, biasM, amask, out);
}
